// GraphAttention_15539191677422
// MI455X (gfx1250) — matrix operand order repaired and checked by reading, not compiled, not run on hardware
//
#include <hip/hip_runtime.h>
#include <hip/hip_bf16.h>

typedef __attribute__((ext_vector_type(16))) _Float16 v16h;
typedef __attribute__((ext_vector_type(8)))  _Float16 v8h;
typedef __attribute__((ext_vector_type(8)))  float    v8f;

#define NN   4096
#define FIN  512
#define FOUT 256
#define NH   8
#define NEGV (-1.0e10f)

__device__ __forceinline__ float lrelu(float x) { return fmaxf(x, 0.2f * x); }

// ---------------------------------------------------------------- cast X -> f16
__global__ void k_cast_x(const float* __restrict__ X, _Float16* __restrict__ Xh) {
  int i = blockIdx.x * 256 + threadIdx.x;
  Xh[i] = (_Float16)X[i];
}

// ------------------------------------------- cast W [h][f][o] -> WhT f16 [h][o][f]
__global__ void k_cast_wt(const float* __restrict__ W, _Float16* __restrict__ WhT) {
  int i = blockIdx.x * 256 + threadIdx.x;      // over H*FIN*FOUT
  int h = i / (FIN * FOUT);
  int r = i % (FIN * FOUT);
  int f = r / FOUT, o = r % FOUT;
  WhT[((size_t)h * FOUT + o) * FIN + f] = (_Float16)W[i];
}

// ---------------------------------------------------------------- projection GEMM
// feats[h][n][o] = sum_f X[n][f] * W[h][f][o]; one wave per 16x16 tile, K=512.
__global__ void __launch_bounds__(128) k_proj(const _Float16* __restrict__ Xh,
                                              const _Float16* __restrict__ WhT,
                                              float* __restrict__ featsF,
                                              _Float16* __restrict__ featsHT) {
  int wid  = blockIdx.x * 4 + (threadIdx.x >> 5);
  int lane = threadIdx.x & 31;
  int h    = wid >> 12;           // 4096 tiles per head
  int rem  = wid & 4095;
  int ti   = rem >> 4;            // 0..255 row tile
  int tj   = rem & 15;            // 0..15  col tile
  int nbase = ti * 16, obase = tj * 16;
  int mrow = lane & 15, half = lane >> 4;

  v8f acc = {};
  const _Float16* arow = Xh + (size_t)(nbase + mrow) * FIN;
  const _Float16* brow = WhT + ((size_t)h * FOUT + obase + mrow) * FIN;
  for (int kb = 0; kb < FIN; kb += 32) {
    v16h a = *(const v16h*)(arow + kb + half * 16);
    v16h b = *(const v16h*)(brow + kb + half * 16);
    acc = __builtin_amdgcn_wmma_f32_16x16x32_f16(false, a, false, b, (short)0, acc,
                                                 false, false);
  }
  int mbase = half * 8;
  for (int v = 0; v < 8; ++v)
    featsF[((size_t)h * NN + nbase + mbase + v) * FOUT + obase + mrow] = acc[v];
  v8h hh;
  for (int v = 0; v < 8; ++v) hh[v] = (_Float16)acc[v];
  *(v8h*)(featsHT + ((size_t)h * FOUT + obase + mrow) * NN + nbase + mbase) = hh;
}

// ------------------------------------------------- s[h,n], t[h,n] row reductions
__global__ void k_st(const float* __restrict__ featsF, const float* __restrict__ a_self,
                     const float* __restrict__ a_neigh, float* __restrict__ s,
                     float* __restrict__ t) {
  int row  = blockIdx.x * 8 + (threadIdx.x >> 5);  // h*N + n
  int lane = threadIdx.x & 31;
  int h    = row >> 12;                            // N == 4096
  const float* f  = featsF + (size_t)row * FOUT;
  const float* as = a_self + h * FOUT;
  const float* an = a_neigh + h * FOUT;
  float ss = 0.f, tt = 0.f;
  for (int o = lane; o < FOUT; o += 32) { float v = f[o]; ss += v * as[o]; tt += v * an[o]; }
  for (int off = 16; off; off >>= 1) {
    ss += __shfl_xor(ss, off, 32);
    tt += __shfl_xor(tt, off, 32);
  }
  if (lane == 0) { s[row] = ss; t[row] = tt; }
}

// --------------------------------------------- flash-style attention + aggregation
// One block per 16-row i-block; 8 waves; loop over heads; double softmax exactly.
__global__ void __launch_bounds__(256) k_attn(const float* __restrict__ A,
                                              const float* __restrict__ s,
                                              const float* __restrict__ t,
                                              const _Float16* __restrict__ featsHT,
                                              const float* __restrict__ bias,
                                              float* __restrict__ out) {
  __shared__ float tLDS[NN];                         // 16 KB
  __shared__ unsigned int Abits[16][128];            // 8 KB, 1 bit per A entry
  __shared__ __align__(64) _Float16 p2tile[16][32];  // 1 KB
  __shared__ float sLDS[16];
  __shared__ float redM[16][16], redZ[16][16];
  __shared__ float rowM1[16], rowI1[16], rowM2[16], rowI2[16];

  int tid  = threadIdx.x;
  int lane = tid & 31;
  int w    = tid >> 5;         // wave id: owns out-cols [32w, 32w+32)
  int i0   = blockIdx.x * 16;

  // Build adjacency bitmask with coalesced loads + wave32 ballot.
  for (int kw = 0; kw < 256; ++kw) {
    int word = w * 256 + kw;                 // 0..2047
    int r = word >> 7, wi = word & 127;
    float av = A[(size_t)(i0 + r) * NN + wi * 32 + lane];
    unsigned long long m = __ballot(av != 0.0f);
    if (lane == 0) Abits[r][wi] = (unsigned int)m;
  }
  __syncthreads();

  v8f acc0 = {}, acc1 = {};
  for (int h = 0; h < NH; ++h) {
    for (int k = tid; k < NN; k += 256) tLDS[k] = t[(size_t)h * NN + k];
    if (tid < 16) sLDS[tid] = s[(size_t)h * NN + i0 + tid];
    __syncthreads();

    int r = tid >> 4, g = tid & 15;
    float sv = sLDS[r];

    // ---- pass 1: online max/sum of e = lrelu(s+t) + mask
    float m = -INFINITY, Z = 0.f;
    for (int j = g * 256; j < g * 256 + 256; ++j) {
      unsigned int bit = (Abits[r][j >> 5] >> (j & 31)) & 1u;
      float e = lrelu(sv + tLDS[j]) + (bit ? 0.f : NEGV);
      float nm = fmaxf(m, e);
      Z = Z * __expf(m - nm) + __expf(e - nm);
      m = nm;
    }
    redM[r][g] = m; redZ[r][g] = Z;
    __syncthreads();
    if (g == 0) {
      float mm = -INFINITY;
      for (int q = 0; q < 16; ++q) mm = fmaxf(mm, redM[r][q]);
      float zz = 0.f;
      for (int q = 0; q < 16; ++q) zz += redZ[r][q] * __expf(redM[r][q] - mm);
      rowM1[r] = mm; rowI1[r] = 1.f / zz;
    }
    __syncthreads();
    float m1 = rowM1[r], i1 = rowI1[r];

    // ---- pass 2: f = p1 + mask; online max/sum
    m = -INFINITY; Z = 0.f;
    for (int j = g * 256; j < g * 256 + 256; ++j) {
      unsigned int bit = (Abits[r][j >> 5] >> (j & 31)) & 1u;
      float f = bit ? (__expf(lrelu(sv + tLDS[j]) - m1) * i1) : NEGV;
      float nm = fmaxf(m, f);
      Z = Z * __expf(m - nm) + __expf(f - nm);
      m = nm;
    }
    redM[r][g] = m; redZ[r][g] = Z;
    __syncthreads();
    if (g == 0) {
      float mm = -INFINITY;
      for (int q = 0; q < 16; ++q) mm = fmaxf(mm, redM[r][q]);
      float zz = 0.f;
      for (int q = 0; q < 16; ++q) zz += redZ[r][q] * __expf(redM[r][q] - mm);
      rowM2[r] = mm; rowI2[r] = 1.f / zz;
    }
    __syncthreads();
    float m2 = rowM2[r], i2 = rowI2[r];

    // ---- pass 3: per 32-wide K chunk, build p2 tile then WMMA-aggregate
    for (int kc = 0; kc < NN / 32; ++kc) {
      int jb = kc * 32;
      for (int u = 0; u < 2; ++u) {
        int c = g * 2 + u;
        int j = jb + c;
        unsigned int bit = (Abits[r][j >> 5] >> (j & 31)) & 1u;
        float p2 = 0.f;
        if (bit) {
          float p1 = __expf(lrelu(sv + tLDS[j]) - m1) * i1;
          p2 = __expf(p1 - m2) * i2;
        }
        p2tile[r][c] = (_Float16)p2;
      }
      __syncthreads();

      int mrow = lane & 15, half = lane >> 4;
      v16h a = *(const v16h*)(&p2tile[mrow][half * 16]);
      const _Float16* bptr =
          featsHT + ((size_t)h * FOUT + w * 32 + mrow) * NN + jb + half * 16;
      v16h b0 = *(const v16h*)(bptr);
      v16h b1 = *(const v16h*)(bptr + (size_t)16 * NN);
      acc0 = __builtin_amdgcn_wmma_f32_16x16x32_f16(false, a, false, b0, (short)0, acc0,
                                                    false, false);
      acc1 = __builtin_amdgcn_wmma_f32_16x16x32_f16(false, a, false, b1, (short)0, acc1,
                                                    false, false);
      __syncthreads();
    }
  }

  // ---- epilogue: mean over heads, + mean bias, relu
  int mrow = lane & 15, half = lane >> 4;
  int c0 = w * 32 + mrow, c1 = c0 + 16;
  float b0s = 0.f, b1s = 0.f;
  for (int h = 0; h < NH; ++h) { b0s += bias[h * FOUT + c0]; b1s += bias[h * FOUT + c1]; }
  for (int v = 0; v < 8; ++v) {
    int rr = i0 + half * 8 + v;
    out[(size_t)rr * FOUT + c0] = fmaxf((acc0[v] + b0s) * (1.f / NH), 0.f);
    out[(size_t)rr * FOUT + c1] = fmaxf((acc1[v] + b1s) * (1.f / NH), 0.f);
  }
}

extern "C" void kernel_launch(void* const* d_in, const int* in_sizes, int n_in,
                              void* d_out, int out_size, void* d_ws, size_t ws_size,
                              hipStream_t stream) {
  const float* X       = (const float*)d_in[0];
  const float* A       = (const float*)d_in[1];
  const float* W       = (const float*)d_in[2];
  const float* b       = (const float*)d_in[3];
  const float* a_self  = (const float*)d_in[4];
  const float* a_neigh = (const float*)d_in[5];
  float* out = (float*)d_out;

  char* ws = (char*)d_ws;
  const size_t MB = 1024u * 1024u;
  _Float16* Xh      = (_Float16*)(ws);                  //  4 MB: [N][FIN]
  _Float16* WhT     = (_Float16*)(ws + 4 * MB);         //  2 MB: [H][FOUT][FIN]
  float*    featsF  = (float*)   (ws + 6 * MB);         // 32 MB: [H][N][FOUT]
  _Float16* featsHT = (_Float16*)(ws + 38 * MB);        // 16 MB: [H][FOUT][N]
  float*    s       = (float*)   (ws + 54 * MB);        // 128 KB: [H][N]
  float*    t       = (float*)   (ws + 54 * MB + 128 * 1024);  // 128 KB: [H][N]

  k_cast_x <<<(NN * FIN) / 256, 256, 0, stream>>>(X, Xh);
  k_cast_wt<<<(NH * FIN * FOUT) / 256, 256, 0, stream>>>(W, WhT);
  k_proj   <<<(NH * (NN / 16) * (FOUT / 16)) / 4, 128, 0, stream>>>(Xh, WhT, featsF, featsHT);
  k_st     <<<(NH * NN) / 8, 256, 0, stream>>>(featsF, a_self, a_neigh, s, t);
  k_attn   <<<NN / 16, 256, 0, stream>>>(A, s, t, featsHT, b, out);
}
